// BiMambaBlock_62457414418902
// MI455X (gfx1250) — compile-verified, matrix operand order checked by reading, not run on hardware
//
#include <hip/hip_runtime.h>

typedef __bf16 bf16_t;
typedef __attribute__((ext_vector_type(16))) __bf16 v16bf;
typedef __attribute__((ext_vector_type(8)))  __bf16 v8bf;
typedef __attribute__((ext_vector_type(8)))  float   v8f;

// GCC-style 128-bit int vector, matching the async-LDS builtin's pointee type
typedef int v4i_vs __attribute__((vector_size(16)));
typedef __attribute__((address_space(1))) v4i_vs* gptr_v4i;
typedef __attribute__((address_space(3))) v4i_vs* lptr_v4i;

#if __has_builtin(__builtin_amdgcn_global_load_async_to_lds_b128) && \
    __has_builtin(__builtin_amdgcn_s_wait_asynccnt)
#define HAVE_ASYNC_LDS 1
#else
#define HAVE_ASYNC_LDS 0
#endif

// ---------- helpers ----------
__device__ __forceinline__ bf16_t f2bf(float f) {
    union { float f; unsigned u; } c; c.f = f;
    unsigned r = c.u + 0x7FFFu + ((c.u >> 16) & 1u);   // RNE
    unsigned short h = (unsigned short)(r >> 16);
    return __builtin_bit_cast(bf16_t, h);
}

__device__ __forceinline__ float sigmoidf_(float x) { return 1.f / (1.f + __expf(-x)); }
__device__ __forceinline__ float siluf_(float x)    { return x * sigmoidf_(x); }
__device__ __forceinline__ float softplusf_(float x) {
    return (x > 20.f) ? x : __logf(1.f + __expf(x));
}

// 16-byte memory->LDS copy: async (ASYNCcnt, bypasses VGPRs) when available
__device__ __forceinline__ void copy16_to_lds(const bf16_t* g, bf16_t* l) {
#if HAVE_ASYNC_LDS
    __builtin_amdgcn_global_load_async_to_lds_b128(
        (gptr_v4i)(bf16_t*)g, (lptr_v4i)l, 0, 0);
#else
    *(v8bf*)l = *(const v8bf*)g;
#endif
}

template <int N>
__device__ __forceinline__ void wait_async_c() {
#if HAVE_ASYNC_LDS
    __builtin_amdgcn_s_wait_asynccnt(N);
#endif
}

// ---------- cast kernels ----------
__global__ void cast_f32_bf16_kernel(const float* __restrict__ in, bf16_t* __restrict__ out, size_t n) {
    size_t i = (size_t)blockIdx.x * blockDim.x + threadIdx.x;
    if (i < n) out[i] = f2bf(in[i]);
}

// x (B,L,Dm) -> bf16 copy + L-flipped bf16 copy
__global__ void cast_x_kernel(const float* __restrict__ x, bf16_t* __restrict__ xb,
                              bf16_t* __restrict__ xbFlip, int Bb, int Ll, int Dm) {
    size_t i = (size_t)blockIdx.x * blockDim.x + threadIdx.x;
    size_t total = (size_t)Bb * Ll * Dm;
    if (i >= total) return;
    int d = (int)(i % Dm);
    int l = (int)((i / Dm) % Ll);
    int b = (int)(i / ((size_t)Dm * Ll));
    bf16_t v = f2bf(x[i]);
    xb[i] = v;
    xbFlip[((size_t)(b * Ll + (Ll - 1 - l))) * Dm + d] = v;
}

// ---------- WMMA GEMM: C[M,N] = A[M,K] * W[N,K]^T, fused epilogue ----------
// Block tile 128x128, 8 wave32 waves, 32x64 per wave (2x4 WMMA accums).
// Requirements: M % 128 == 0, K % 32 == 0 (true for all call sites).
// N may be ragged (x-proj N=96): B-row index is clamped, epilogue guards n.
// Double-buffered LDS staging via GLOBAL_LOAD_ASYNC_TO_LDS_B128 (ASYNCcnt).
// EPI: 0 = store f32, 1 = store f32 + bf16, 2 = bias+softplus -> f32,
//      3 = bias+silu -> bf16, 4 = bias + resid -> f32
template <int EPI>
__global__ __launch_bounds__(256) void wmma_gemm_kernel(
    const bf16_t* __restrict__ A, int lda,
    const bf16_t* __restrict__ W, int ldb,
    int M, int N, int K,
    float* __restrict__ outF, bf16_t* __restrict__ outH, int ldc,
    const float* __restrict__ bias, const float* __restrict__ resid)
{
    __shared__ bf16_t As[2][128 * 32];
    __shared__ bf16_t Bs[2][128 * 32];

    const int tid  = threadIdx.x;
    const int lane = tid & 31;
    const int wave = tid >> 5;       // 0..7
    const int wm   = wave & 3;       // 0..3  -> M offset wm*32
    const int wn   = wave >> 2;      // 0..1  -> N offset wn*64
    const int blockM = blockIdx.y * 128;
    const int blockN = blockIdx.x * 128;

    const int l16  = lane & 15;
    const int koff = (lane >> 4) * 8;

    // staging coordinates (2 chunks each for A and B per thread)
    const int row0 = tid >> 2;            // 0..63
    const int row1 = row0 + 64;           // 64..127
    const int scol = (tid & 3) * 8;       // 0,8,16,24

    v8f acc[2][4];
#pragma unroll
    for (int i = 0; i < 2; ++i)
#pragma unroll
        for (int j = 0; j < 4; ++j)
#pragma unroll
            for (int r = 0; r < 8; ++r) acc[i][j][r] = 0.f;

    const int gm0 = blockM + row0, gm1 = blockM + row1;
    const int gn0 = min(blockN + row0, N - 1);
    const int gn1 = min(blockN + row1, N - 1);

    auto stage = [&](int k0, int buf) {
        copy16_to_lds(A + (size_t)gm0 * lda + k0 + scol, &As[buf][row0 * 32 + scol]);
        copy16_to_lds(A + (size_t)gm1 * lda + k0 + scol, &As[buf][row1 * 32 + scol]);
        copy16_to_lds(W + (size_t)gn0 * ldb + k0 + scol, &Bs[buf][row0 * 32 + scol]);
        copy16_to_lds(W + (size_t)gn1 * ldb + k0 + scol, &Bs[buf][row1 * 32 + scol]);
    };

    const int nk = K >> 5;
    stage(0, 0);

    for (int ks = 0; ks < nk; ++ks) {
        const int cur = ks & 1;
        if (ks + 1 < nk) {
            stage((ks + 1) << 5, cur ^ 1);
            wait_async_c<4>();   // own 4 outstanding = stage ks complete
        } else {
            wait_async_c<0>();
        }
        __syncthreads();

        v16bf afrag[2], bfrag[4];
#pragma unroll
        for (int i = 0; i < 2; ++i) {
            const bf16_t* pA = &As[cur][(wm * 32 + i * 16 + l16) * 32];
            v8bf lo = *(const v8bf*)(pA + koff);
            v8bf hi = *(const v8bf*)(pA + 16 + koff);
            afrag[i] = __builtin_shufflevector(lo, hi, 0,1,2,3,4,5,6,7,8,9,10,11,12,13,14,15);
        }
#pragma unroll
        for (int j = 0; j < 4; ++j) {
            const bf16_t* pB = &Bs[cur][(wn * 64 + j * 16 + l16) * 32];
            v8bf lo = *(const v8bf*)(pB + koff);
            v8bf hi = *(const v8bf*)(pB + 16 + koff);
            bfrag[j] = __builtin_shufflevector(lo, hi, 0,1,2,3,4,5,6,7,8,9,10,11,12,13,14,15);
        }

#pragma unroll
        for (int i = 0; i < 2; ++i)
#pragma unroll
            for (int j = 0; j < 4; ++j)
                acc[i][j] = __builtin_amdgcn_wmma_f32_16x16x32_bf16(
                    false, afrag[i], false, bfrag[j], (short)0, acc[i][j], false, false);

        __syncthreads();   // safe to overwrite buffer `cur` at iteration ks+1's stage
    }

    // ---- epilogue ----
#pragma unroll
    for (int i = 0; i < 2; ++i) {
        const int mBase = blockM + wm * 32 + i * 16 + 8 * (lane >> 4);
#pragma unroll
        for (int j = 0; j < 4; ++j) {
            const int n = blockN + wn * 64 + j * 16 + l16;
            if (n >= N) continue;
#pragma unroll
            for (int r = 0; r < 8; ++r) {
                const int m = mBase + r;
                float v = acc[i][j][r];
                if (EPI == 2) { v = softplusf_(v + bias[n]); }
                else if (EPI == 3) { v = siluf_(v + bias[n]); }
                else if (EPI == 4) { v = v + bias[n] + resid[(size_t)m * ldc + n]; }
                if (EPI == 0 || EPI == 1 || EPI == 2 || EPI == 4)
                    outF[(size_t)m * ldc + n] = v;
                if (EPI == 1 || EPI == 3)
                    outH[(size_t)m * ldc + n] = f2bf(v);
            }
        }
    }
}

// ---------- depthwise causal conv(4) + bias + SiLU ----------
__global__ void conv_silu_kernel(const float* __restrict__ xz, const float* __restrict__ conv_w,
                                 const float* __restrict__ conv_b,
                                 float* __restrict__ xcF, bf16_t* __restrict__ xcH,
                                 int Bb, int Ll, int Din, int ldxz)
{
    size_t i = (size_t)blockIdx.x * blockDim.x + threadIdx.x;
    size_t total = (size_t)Bb * Ll * Din;
    if (i >= total) return;
    int d = (int)(i % Din);
    int l = (int)((i / Din) % Ll);
    int b = (int)(i / ((size_t)Din * Ll));
    float acc = conv_b[d];
#pragma unroll
    for (int j = 0; j < 4; ++j) {
        int ls = l - 3 + j;
        if (ls >= 0)
            acc += conv_w[d * 4 + j] * xz[((size_t)(b * Ll + ls)) * ldxz + d];
    }
    float s = siluf_(acc);
    xcF[i] = s;
    xcH[i] = f2bf(s);
}

// ---------- selective scan: 16 lanes (states) per channel, wave32 shfl reduce ----------
__global__ __launch_bounds__(256) void scan_kernel(
    const float* __restrict__ dtF, const float* __restrict__ xdblF,
    const float* __restrict__ xcF, const float* __restrict__ xzF,
    const float* __restrict__ A_log, const float* __restrict__ Dvec,
    float* __restrict__ yF, bf16_t* __restrict__ yH,
    int Bb, int Ll, int Din, int Dst, int dtRank, int ldxdbl, int ldxz)
{
    int gid = blockIdx.x * blockDim.x + threadIdx.x;
    int s  = gid & 15;
    int ch = gid >> 4;
    if (ch >= Bb * Din) return;
    int d = ch % Din;
    int b = ch / Din;

    float Acoef = -__expf(A_log[d * Dst + s]);
    float Dcoef = Dvec[d];
    float h = 0.f;

    for (int l = 0; l < Ll; ++l) {
        size_t row = (size_t)(b * Ll + l);
        float dt = dtF[row * Din + d];
        float u  = xcF[row * Din + d];
        float Bv = xdblF[row * ldxdbl + dtRank + s];
        float Cv = xdblF[row * ldxdbl + dtRank + Dst + s];
        h = h * __expf(dt * Acoef) + dt * u * Bv;
        float p = h * Cv;
        p += __shfl_xor(p, 8, 32);
        p += __shfl_xor(p, 4, 32);
        p += __shfl_xor(p, 2, 32);
        p += __shfl_xor(p, 1, 32);
        if (s == 0) {
            float z  = xzF[row * ldxz + Din + d];
            float yv = (p + u * Dcoef) * siluf_(z);
            yF[row * Din + d] = yv;
            yH[row * Din + d] = f2bf(yv);
        }
    }
}

// ---------- x1 = x + 0.5*(y_f + flip(y_b)) ----------
__global__ void combine_kernel(const float* __restrict__ x, const float* __restrict__ yf,
                               const float* __restrict__ yb,
                               float* __restrict__ x1F,
                               int Bb, int Ll, int Dm)
{
    size_t i = (size_t)blockIdx.x * blockDim.x + threadIdx.x;
    size_t total = (size_t)Bb * Ll * Dm;
    if (i >= total) return;
    int d = (int)(i % Dm);
    int l = (int)((i / Dm) % Ll);
    int b = (int)(i / ((size_t)Dm * Ll));
    size_t bwd = ((size_t)(b * Ll + (Ll - 1 - l))) * Dm + d;
    float v = x[i] + 0.5f * (yf[i] + yb[bwd]);
    x1F[i] = v;
}

// ---------- row LayerNorm -> bf16 ----------
__global__ __launch_bounds__(256) void layernorm_kernel(
    const float* __restrict__ xin, const float* __restrict__ g, const float* __restrict__ bta,
    bf16_t* __restrict__ outH, int Dm)
{
    int row = blockIdx.x;
    const float* px = xin + (size_t)row * Dm;
    float sum = 0.f, sumsq = 0.f;
    for (int i = threadIdx.x; i < Dm; i += 256) {
        float v = px[i]; sum += v; sumsq += v * v;
    }
#pragma unroll
    for (int o = 16; o >= 1; o >>= 1) {
        sum   += __shfl_xor(sum, o, 32);
        sumsq += __shfl_xor(sumsq, o, 32);
    }
    __shared__ float s1[8], s2[8];
    int w = threadIdx.x >> 5, ln = threadIdx.x & 31;
    if (ln == 0) { s1[w] = sum; s2[w] = sumsq; }
    __syncthreads();
    if (threadIdx.x == 0) {
        float a = 0.f, q = 0.f;
        for (int i = 0; i < 8; ++i) { a += s1[i]; q += s2[i]; }
        s1[0] = a; s2[0] = q;
    }
    __syncthreads();
    float mean = s1[0] / Dm;
    float var  = s2[0] / Dm - mean * mean;
    float inv  = rsqrtf(var + 1e-5f);
    for (int i = threadIdx.x; i < Dm; i += 256) {
        float v = (px[i] - mean) * inv * g[i] + bta[i];
        outH[(size_t)row * Dm + i] = f2bf(v);
    }
}

// ---------- host ----------
template <int EPI>
static void launch_gemm(const bf16_t* A, int lda, const bf16_t* W, int ldb,
                        int M, int N, int K,
                        float* outF, bf16_t* outH, int ldc,
                        const float* bias, const float* resid, hipStream_t s)
{
    dim3 grid((N + 127) / 128, (M + 127) / 128);
    wmma_gemm_kernel<EPI><<<grid, 256, 0, s>>>(A, lda, W, ldb, M, N, K, outF, outH, ldc, bias, resid);
}

static void launch_cast(const float* in, bf16_t* out, size_t n, hipStream_t s)
{
    cast_f32_bf16_kernel<<<(unsigned)((n + 255) / 256), 256, 0, s>>>(in, out, n);
}

extern "C" void kernel_launch(void* const* d_in, const int* in_sizes, int n_in,
                              void* d_out, int out_size, void* d_ws, size_t ws_size,
                              hipStream_t stream)
{
    const int Bb = 2, Ll = 2048, Dm = 1024, Din = 2048, Dst = 16, DtR = 64;
    const int Mrows = Bb * Ll;                 // 4096
    const int NXdbl = DtR + 2 * Dst;           // 96

    const float* x = (const float*)d_in[0];
    const float* ff_ln_g = (const float*)d_in[19];
    const float* ff_ln_b = (const float*)d_in[20];
    const float* ff_w1   = (const float*)d_in[21];
    const float* ff_b1   = (const float*)d_in[22];
    const float* ff_w2   = (const float*)d_in[23];
    const float* ff_b2   = (const float*)d_in[24];

    char* p = (char*)d_ws;
    auto alloc = [&](size_t bytes) -> void* {
        void* r = (void*)p;
        p += (bytes + 255) & ~(size_t)255;
        return r;
    };

    bf16_t* xbH     = (bf16_t*)alloc((size_t)Mrows * Dm * 2);
    bf16_t* xbFlipH = (bf16_t*)alloc((size_t)Mrows * Dm * 2);
    bf16_t* in_wH   = (bf16_t*)alloc((size_t)2 * Din * Dm * 2);
    bf16_t* x_wH    = (bf16_t*)alloc((size_t)NXdbl * Din * 2);
    bf16_t* dt_wH   = (bf16_t*)alloc((size_t)Din * DtR * 2);
    bf16_t* out_wH  = (bf16_t*)alloc((size_t)Dm * Din * 2);
    bf16_t* ffw1H   = (bf16_t*)alloc((size_t)4 * Dm * Dm * 2);
    bf16_t* ffw2H   = (bf16_t*)alloc((size_t)Dm * 4 * Dm * 2);

    float*  xzF   = (float*)alloc((size_t)Mrows * 2 * Din * 4);
    float*  xcF   = (float*)alloc((size_t)Mrows * Din * 4);
    bf16_t* xcH   = (bf16_t*)alloc((size_t)Mrows * Din * 2);
    float*  xdblF = (float*)alloc((size_t)Mrows * NXdbl * 4);
    bf16_t* xdblH = (bf16_t*)alloc((size_t)Mrows * NXdbl * 2);
    float*  dtF   = (float*)alloc((size_t)Mrows * Din * 4);
    float*  yF    = (float*)alloc((size_t)Mrows * Din * 4);
    bf16_t* yH    = (bf16_t*)alloc((size_t)Mrows * Din * 2);
    float*  yfF   = (float*)alloc((size_t)Mrows * Dm * 4);
    float*  ybF   = (float*)alloc((size_t)Mrows * Dm * 4);
    float*  x1F   = (float*)alloc((size_t)Mrows * Dm * 4);
    bf16_t* lnH   = (bf16_t*)alloc((size_t)Mrows * Dm * 2);
    bf16_t* hH    = (bf16_t*)alloc((size_t)Mrows * 4 * Dm * 2);
    (void)ws_size; (void)in_sizes; (void)n_in;

    // x -> bf16 (+flipped copy for backward direction)
    {
        size_t n = (size_t)Mrows * Dm;
        cast_x_kernel<<<(unsigned)((n + 255) / 256), 256, 0, stream>>>(x, xbH, xbFlipH, Bb, Ll, Dm);
    }

    // FFN weights -> bf16 (direction-independent)
    launch_cast(ff_w1, ffw1H, (size_t)4 * Dm * Dm, stream);
    launch_cast(ff_w2, ffw2H, (size_t)Dm * 4 * Dm, stream);

    for (int dir = 0; dir < 2; ++dir) {
        const float* in_w   = (const float*)d_in[1 + dir * 9 + 0];
        const float* conv_w = (const float*)d_in[1 + dir * 9 + 1];
        const float* conv_b = (const float*)d_in[1 + dir * 9 + 2];
        const float* x_w    = (const float*)d_in[1 + dir * 9 + 3];
        const float* dt_w   = (const float*)d_in[1 + dir * 9 + 4];
        const float* dt_b   = (const float*)d_in[1 + dir * 9 + 5];
        const float* A_log  = (const float*)d_in[1 + dir * 9 + 6];
        const float* Dvec   = (const float*)d_in[1 + dir * 9 + 7];
        const float* out_w  = (const float*)d_in[1 + dir * 9 + 8];

        const bf16_t* xsrc = (dir == 0) ? xbH : xbFlipH;
        float* ydst = (dir == 0) ? yfF : ybF;

        // weights -> bf16 into shared staging buffers (dirs run sequentially)
        launch_cast(in_w,  in_wH,  (size_t)2 * Din * Dm, stream);
        launch_cast(x_w,   x_wH,   (size_t)NXdbl * Din,  stream);
        launch_cast(dt_w,  dt_wH,  (size_t)Din * DtR,    stream);
        launch_cast(out_w, out_wH, (size_t)Dm * Din,     stream);

        // 1) xz = x @ in_w^T   [4096 x 4096]
        launch_gemm<0>(xsrc, Dm, in_wH, Dm, Mrows, 2 * Din, Dm,
                       xzF, nullptr, 2 * Din, nullptr, nullptr, stream);

        // 2) depthwise conv + SiLU
        {
            size_t n = (size_t)Mrows * Din;
            conv_silu_kernel<<<(unsigned)((n + 255) / 256), 256, 0, stream>>>(
                xzF, conv_w, conv_b, xcF, xcH, Bb, Ll, Din, 2 * Din);
        }

        // 3) xdbl = xc @ x_w^T   [4096 x 96]
        launch_gemm<1>(xcH, Din, x_wH, Din, Mrows, NXdbl, Din,
                       xdblF, xdblH, NXdbl, nullptr, nullptr, stream);

        // 4) dt = softplus(xdbl[:, :64] @ dt_w^T + dt_b)   [4096 x 2048]
        launch_gemm<2>(xdblH, NXdbl, dt_wH, DtR, Mrows, Din, DtR,
                       dtF, nullptr, Din, dt_b, nullptr, stream);

        // 5) selective scan (fused +xc*D and *silu(z))
        {
            int nthreads = Bb * Din * Dst;   // 65536
            scan_kernel<<<nthreads / 256, 256, 0, stream>>>(
                dtF, xdblF, xcF, xzF, A_log, Dvec, yF, yH,
                Bb, Ll, Din, Dst, DtR, NXdbl, 2 * Din);
        }

        // 6) y_out = y @ out_w^T   [4096 x 1024]
        launch_gemm<0>(yH, Din, out_wH, Din, Mrows, Dm, Din,
                       ydst, nullptr, Dm, nullptr, nullptr, stream);
    }

    // x1 = x + 0.5*(y_f + flip(y_b))
    {
        size_t n = (size_t)Mrows * Dm;
        combine_kernel<<<(unsigned)((n + 255) / 256), 256, 0, stream>>>(
            x, yfF, ybF, x1F, Bb, Ll, Dm);
    }

    // LayerNorm -> bf16
    layernorm_kernel<<<Mrows, 256, 0, stream>>>(x1F, ff_ln_g, ff_ln_b, lnH, Dm);

    // FFN1: h = silu(ln @ ff_w1^T + b1)   [4096 x 4096]
    launch_gemm<3>(lnH, Dm, ffw1H, Dm, Mrows, 4 * Dm, Dm,
                   nullptr, hH, 4 * Dm, ff_b1, nullptr, stream);

    // FFN2: out = x1 + (h @ ff_w2^T + b2)   [4096 x 1024] -> d_out (fp32)
    launch_gemm<4>(hH, 4 * Dm, ffw2H, 4 * Dm, Mrows, Dm, 4 * Dm,
                   (float*)d_out, nullptr, Dm, ff_b2, x1F, stream);
    (void)out_size;
}
